// TrajectorySurrogate_12249246728961
// MI455X (gfx1250) — matrix operand order repaired and checked by reading, not compiled, not run on hardware
//
#include <hip/hip_runtime.h>

// ---------------------------------------------------------------------------
// TrajectorySurrogate on MI455X (gfx1250, wave32, WMMA)
// B=8192, IN=7, H=256, OUT=3, T=200
//
// Strategy: batch rows are independent through the recurrence, so each block
// owns 32 rows and runs all 200 steps with LSTM state resident in LDS.
// Big GEMMs use v_wmma_f32_16x16x32_bf16 (A from LDS, B from L2-resident
// bf16 weights pre-converted into d_ws). Tiny-K layers use VALU.
// ---------------------------------------------------------------------------

#define H_       256
#define ROWS     32
#define TSTEPS   200
#define OUTD     3
#define NTHREADS 512    // 16 wave32s

#define HB  264   // bf16 LDS row stride (8-elem pad -> conflict-free b128)
#define CF  260   // f32 LDS row stride for cell state
#define OF  132   // f32 LDS row stride for out-head intermediate

typedef __attribute__((ext_vector_type(16))) __bf16 bf16x16;
typedef __attribute__((ext_vector_type(8)))  float  f32x8;
typedef __attribute__((ext_vector_type(4)))  unsigned u32x4;

__device__ __forceinline__ bf16x16 pack_frag(u32x4 a, u32x4 b) {
  union { u32x4 q[2]; bf16x16 f; } u;
  u.q[0] = a; u.q[1] = b;
  return u.f;
}

// A fragment (16x32 bf16, MxK): lanes 0-15 hold M=lane, K=[k0..k0+7] in v0-3
// and K=[k0+16..k0+23] in v4-7; lanes 16-31 hold K offsets +8 / +24.
__device__ __forceinline__ bf16x16 frag_a_lds(const __bf16* base, int m, int k0, int lane) {
  int row = m * 16 + (lane & 15);
  int col = k0 + ((lane >> 4) << 3);            // +0 or +8
  const __bf16* p = base + row * HB + col;
  return pack_frag(*(const u32x4*)(p), *(const u32x4*)(p + 16));
}

// B fragment (32x16 bf16, KxN) from row-major W (N x K): lane n holds
// K=[k0..k0+7] and K=[k0+16..k0+23]; lanes 16-31 hold K offsets +8 / +24.
__device__ __forceinline__ bf16x16 frag_b_glb(const __bf16* W, int nbase, int k0, int lane, int K) {
  int n   = nbase + (lane & 15);
  int col = k0 + ((lane >> 4) << 3);            // +0 or +8
  const __bf16* p = W + n * K + col;
  return pack_frag(*(const u32x4*)(p), *(const u32x4*)(p + 16));
}

__device__ __forceinline__ f32x8 wmma_bf16(bf16x16 a, bf16x16 b, f32x8 c) {
  return __builtin_amdgcn_wmma_f32_16x16x32_bf16(false, a, false, b, (short)0, c, false, false);
}

__device__ __forceinline__ float sig_(float x)  { return 1.0f / (1.0f + __expf(-x)); }
__device__ __forceinline__ float tanh_(float x) { float e = __expf(-2.0f * x); return (1.0f - e) / (1.0f + e); }

// Per-wave: 2 M-tiles x 1 N-block dense matmul over K
__device__ __forceinline__ void mm_2m(const __bf16* A, const __bf16* W, int K,
                                      int nbase, int lane, f32x8& c0, f32x8& c1) {
  for (int k = 0; k < K; k += 32) {
    bf16x16 a0 = frag_a_lds(A, 0, k, lane);
    bf16x16 a1 = frag_a_lds(A, 1, k, lane);
    bf16x16 b  = frag_b_glb(W, nbase, k, lane, K);
    c0 = wmma_bf16(a0, b, c0);
    c1 = wmma_bf16(a1, b, c1);
  }
}

// --------------------------- weight fp32->bf16 ------------------------------
extern "C" __global__ void cvt_f32_bf16(const float* __restrict__ src,
                                        __bf16* __restrict__ dst, int n) {
  int i = blockIdx.x * blockDim.x + threadIdx.x;
  if (i < n) dst[i] = (__bf16)src[i];
}

// ------------------------------- main kernel --------------------------------
#define LDS_BYTES 158208

extern "C" __global__ __launch_bounds__(NTHREADS)
void traj_main(const float* __restrict__ x,       const float* __restrict__ enc_w1,
               const float* __restrict__ enc_b1,  const float* __restrict__ enc_b2,
               const float* __restrict__ enc_b3,  const float* __restrict__ hinit_b,
               const float* __restrict__ cinit_b, const float* __restrict__ w_ih0,
               const float* __restrict__ b_ih0,   const float* __restrict__ b_hh0,
               const float* __restrict__ b_ih1,   const float* __restrict__ b_hh1,
               const float* __restrict__ out_b1,  const float* __restrict__ out_w2,
               const float* __restrict__ out_b2,  const float* __restrict__ start_token,
               const __bf16* __restrict__ w2_bf,   const __bf16* __restrict__ w3_bf,
               const __bf16* __restrict__ hi_bf,   const __bf16* __restrict__ ci_bf,
               const __bf16* __restrict__ whh0_bf, const __bf16* __restrict__ wih1_bf,
               const __bf16* __restrict__ whh1_bf, const __bf16* __restrict__ ow1_bf,
               float* __restrict__ out) {
  extern __shared__ char smem[];
  __bf16* sH0  = (__bf16*)(smem + 0);        // 32x264 bf16
  __bf16* sH1  = (__bf16*)(smem + 16896);
  __bf16* sE   = (__bf16*)(smem + 33792);
  float*  sC0  = (float*) (smem + 50688);    // 32x260 f32
  float*  sC1  = (float*) (smem + 83968);
  float*  sO1  = (float*) (smem + 117248);   // 32x132 f32 (head intermediate)
  __bf16* sEB  = (__bf16*)(smem + 117248);   // alias: encoder ping buffer
  float*  sDec = (float*) (smem + 134144);   // 32x4
  float*  sWi0 = (float*) (smem + 134656);   // 1024x3
  float*  sB0  = (float*) (smem + 146944);   // 1024 (b_ih0+b_hh0)
  float*  sB1  = (float*) (smem + 151040);   // 1024
  float*  sOb1 = (float*) (smem + 155136);   // 128
  float*  sW2o = (float*) (smem + 155648);   // 3x128
  float*  sX   = (float*) (smem + 157184);   // 32x8

  const int tid  = threadIdx.x;
  const int wave = tid >> 5;
  const int lane = tid & 31;
  const int hi8  = (lane >> 4) << 3;   // C/D layout: lanes 16-31 -> rows +8
  const int lcol = lane & 15;
  const long row0 = (long)blockIdx.x * ROWS;

  // ---- stage small constants & inputs into LDS ----
  for (int i = tid; i < 1024 * 3; i += NTHREADS) sWi0[i] = w_ih0[i];
  for (int i = tid; i < 1024;     i += NTHREADS) sB0[i]  = b_ih0[i] + b_hh0[i];
  for (int i = tid; i < 1024;     i += NTHREADS) sB1[i]  = b_ih1[i] + b_hh1[i];
  for (int i = tid; i < 128;      i += NTHREADS) sOb1[i] = out_b1[i];
  for (int i = tid; i < 384;      i += NTHREADS) sW2o[i] = out_w2[i];
  for (int i = tid; i < ROWS * 8; i += NTHREADS) {
    int r = i >> 3, c = i & 7;
    sX[i] = (c < 7) ? x[(row0 + r) * 7 + c] : 0.0f;
  }
  for (int i = tid; i < ROWS * 4; i += NTHREADS) {
    int c = i & 3;
    sDec[i] = (c < 3) ? start_token[c] : 0.0f;
  }
  __syncthreads();

  // ---- encoder layer 1 (K=7 -> VALU) ----
  for (int i = tid; i < ROWS * H_; i += NTHREADS) {
    int r = i >> 8, c = i & 255;
    float s = enc_b1[c];
#pragma unroll
    for (int j = 0; j < 7; j++) s += sX[r * 8 + j] * enc_w1[c * 7 + j];
    sE[r * HB + c] = (__bf16)fmaxf(s, 0.0f);
  }
  __syncthreads();

  // ---- encoder layer 2 (WMMA) ----
  {
    f32x8 c0 = {}, c1 = {};
    mm_2m(sE, w2_bf, 256, wave * 16, lane, c0, c1);
#pragma unroll
    for (int m = 0; m < 2; m++) {
      const f32x8& a = m ? c1 : c0;
#pragma unroll
      for (int r = 0; r < 8; r++) {
        int row = m * 16 + r + hi8, col = wave * 16 + lcol;
        sEB[row * HB + col] = (__bf16)fmaxf(a[r] + enc_b2[col], 0.0f);
      }
    }
  }
  __syncthreads();

  // ---- encoder layer 3 (WMMA) ----
  {
    f32x8 c0 = {}, c1 = {};
    mm_2m(sEB, w3_bf, 256, wave * 16, lane, c0, c1);
#pragma unroll
    for (int m = 0; m < 2; m++) {
      const f32x8& a = m ? c1 : c0;
#pragma unroll
      for (int r = 0; r < 8; r++) {
        int row = m * 16 + r + hi8, col = wave * 16 + lcol;
        sE[row * HB + col] = (__bf16)fmaxf(a[r] + enc_b3[col], 0.0f);
      }
    }
  }
  __syncthreads();

  // ---- h/c state init: e @ hinit_w^T, e @ cinit_w^T (512 cols each) ----
#pragma unroll
  for (int half = 0; half < 2; half++) {
    int nbase = (wave + half * 16) * 16;   // 0..496
    f32x8 h0a = {}, h1a = {}, c0a = {}, c1a = {};
    for (int k = 0; k < 256; k += 32) {
      bf16x16 a0 = frag_a_lds(sE, 0, k, lane);
      bf16x16 a1 = frag_a_lds(sE, 1, k, lane);
      bf16x16 bh = frag_b_glb(hi_bf, nbase, k, lane, 256);
      bf16x16 bc = frag_b_glb(ci_bf, nbase, k, lane, 256);
      h0a = wmma_bf16(a0, bh, h0a);  h1a = wmma_bf16(a1, bh, h1a);
      c0a = wmma_bf16(a0, bc, c0a);  c1a = wmma_bf16(a1, bc, c1a);
    }
#pragma unroll
    for (int m = 0; m < 2; m++) {
      const f32x8& hv = m ? h1a : h0a;
      const f32x8& cv = m ? c1a : c0a;
#pragma unroll
      for (int r = 0; r < 8; r++) {
        int row = m * 16 + r + hi8, col = nbase + lcol;
        float hval = hv[r] + hinit_b[col];
        float cval = cv[r] + cinit_b[col];
        if (col < 256) { sH0[row * HB + col] = (__bf16)hval; sC0[row * CF + col] = cval; }
        else           { sH1[row * HB + col - 256] = (__bf16)hval; sC1[row * CF + col - 256] = cval; }
      }
    }
  }
  __syncthreads();

  // ============================ time loop ============================
  for (int t = 0; t < TSTEPS; t++) {
    f32x8 acc[4][2];

    // ----- LSTM layer 0: gates = dec@w_ih0^T + b + h0@w_hh0^T -----
#pragma unroll
    for (int g = 0; g < 4; g++) {
      int col = g * 256 + wave * 16 + lcol;
      float bb = sB0[col];
      float u0 = sWi0[col * 3 + 0], u1 = sWi0[col * 3 + 1], u2 = sWi0[col * 3 + 2];
#pragma unroll
      for (int m = 0; m < 2; m++) {
        f32x8 a;
#pragma unroll
        for (int r = 0; r < 8; r++) {
          int row = m * 16 + r + hi8;
          a[r] = bb + sDec[row * 4 + 0] * u0 + sDec[row * 4 + 1] * u1 + sDec[row * 4 + 2] * u2;
        }
        acc[g][m] = a;
      }
    }
    for (int k = 0; k < 256; k += 32) {
      bf16x16 a0 = frag_a_lds(sH0, 0, k, lane);
      bf16x16 a1 = frag_a_lds(sH0, 1, k, lane);
#pragma unroll
      for (int g = 0; g < 4; g++) {
        bf16x16 b = frag_b_glb(whh0_bf, g * 256 + wave * 16, k, lane, 256);
        acc[g][0] = wmma_bf16(a0, b, acc[g][0]);
        acc[g][1] = wmma_bf16(a1, b, acc[g][1]);
      }
    }
    __syncthreads();   // all reads of h0/dec complete
#pragma unroll
    for (int m = 0; m < 2; m++) {
#pragma unroll
      for (int r = 0; r < 8; r++) {
        int row = m * 16 + r + hi8, col = wave * 16 + lcol;
        float iv = acc[0][m][r], fv = acc[1][m][r], gv = acc[2][m][r], ov = acc[3][m][r];
        float cn = sig_(fv) * sC0[row * CF + col] + sig_(iv) * tanh_(gv);
        sC0[row * CF + col] = cn;
        sH0[row * HB + col] = (__bf16)(sig_(ov) * tanh_(cn));
      }
    }
    __syncthreads();   // h0n visible

    // ----- LSTM layer 1: gates = h0n@w_ih1^T + b + h1@w_hh1^T -----
#pragma unroll
    for (int g = 0; g < 4; g++) {
      float bb = sB1[g * 256 + wave * 16 + lcol];
      f32x8 a;
#pragma unroll
      for (int r = 0; r < 8; r++) a[r] = bb;
      acc[g][0] = a; acc[g][1] = a;
    }
    for (int k = 0; k < 256; k += 32) {
      bf16x16 x0 = frag_a_lds(sH0, 0, k, lane);
      bf16x16 x1 = frag_a_lds(sH0, 1, k, lane);
      bf16x16 h0f = frag_a_lds(sH1, 0, k, lane);
      bf16x16 h1f = frag_a_lds(sH1, 1, k, lane);
#pragma unroll
      for (int g = 0; g < 4; g++) {
        int nb = g * 256 + wave * 16;
        bf16x16 bi = frag_b_glb(wih1_bf, nb, k, lane, 256);
        acc[g][0] = wmma_bf16(x0, bi, acc[g][0]);
        acc[g][1] = wmma_bf16(x1, bi, acc[g][1]);
        bf16x16 bh = frag_b_glb(whh1_bf, nb, k, lane, 256);
        acc[g][0] = wmma_bf16(h0f, bh, acc[g][0]);
        acc[g][1] = wmma_bf16(h1f, bh, acc[g][1]);
      }
    }
    __syncthreads();
#pragma unroll
    for (int m = 0; m < 2; m++) {
#pragma unroll
      for (int r = 0; r < 8; r++) {
        int row = m * 16 + r + hi8, col = wave * 16 + lcol;
        float iv = acc[0][m][r], fv = acc[1][m][r], gv = acc[2][m][r], ov = acc[3][m][r];
        float cn = sig_(fv) * sC1[row * CF + col] + sig_(iv) * tanh_(gv);
        sC1[row * CF + col] = cn;
        sH1[row * HB + col] = (__bf16)(sig_(ov) * tanh_(cn));
      }
    }
    __syncthreads();   // h1n visible

    // ----- output head: relu(h1n @ out_w1^T + b1) (32x128, WMMA) -----
    {
      int nb = (wave & 7) * 16;
      int mo = wave >> 3;
      f32x8 o = {};
      for (int k = 0; k < 256; k += 32) {
        bf16x16 a = frag_a_lds(sH1, mo, k, lane);
        bf16x16 b = frag_b_glb(ow1_bf, nb, k, lane, 256);
        o = wmma_bf16(a, b, o);
      }
#pragma unroll
      for (int r = 0; r < 8; r++) {
        int row = mo * 16 + r + hi8, col = nb + lcol;
        sO1[row * OF + col] = fmaxf(o[r] + sOb1[col], 0.0f);
      }
    }
    __syncthreads();

    // ----- final projection to OUT=3 (VALU) + feed back as dec -----
    if (tid < ROWS * OUTD) {
      int r = tid / OUTD, j = tid % OUTD;
      float s = out_b2[j];
      for (int k = 0; k < 128; k++) s += sO1[r * OF + k] * sW2o[j * 128 + k];
      out[((row0 + r) * TSTEPS + t) * OUTD + j] = s;
      sDec[r * 4 + j] = s;
    }
    __syncthreads();
  }
}

// ------------------------------- launcher -----------------------------------
extern "C" void kernel_launch(void* const* d_in, const int* in_sizes, int n_in,
                              void* d_out, int out_size, void* d_ws, size_t ws_size,
                              hipStream_t stream) {
  const float* x       = (const float*)d_in[0];
  const float* enc_w1  = (const float*)d_in[1];
  const float* enc_b1  = (const float*)d_in[2];
  const float* enc_w2  = (const float*)d_in[3];
  const float* enc_b2  = (const float*)d_in[4];
  const float* enc_w3  = (const float*)d_in[5];
  const float* enc_b3  = (const float*)d_in[6];
  const float* hinit_w = (const float*)d_in[7];
  const float* hinit_b = (const float*)d_in[8];
  const float* cinit_w = (const float*)d_in[9];
  const float* cinit_b = (const float*)d_in[10];
  const float* w_ih0   = (const float*)d_in[11];
  const float* w_hh0   = (const float*)d_in[12];
  const float* b_ih0   = (const float*)d_in[13];
  const float* b_hh0   = (const float*)d_in[14];
  const float* w_ih1   = (const float*)d_in[15];
  const float* w_hh1   = (const float*)d_in[16];
  const float* b_ih1   = (const float*)d_in[17];
  const float* b_hh1   = (const float*)d_in[18];
  const float* out_w1  = (const float*)d_in[19];
  const float* out_b1  = (const float*)d_in[20];
  const float* out_w2  = (const float*)d_in[21];
  const float* out_b2  = (const float*)d_in[22];
  const float* start   = (const float*)d_in[23];

  __bf16* ws      = (__bf16*)d_ws;
  __bf16* w2_bf   = ws + 0;
  __bf16* w3_bf   = ws + 65536;
  __bf16* hi_bf   = ws + 131072;
  __bf16* ci_bf   = ws + 262144;
  __bf16* whh0_bf = ws + 393216;
  __bf16* wih1_bf = ws + 655360;
  __bf16* whh1_bf = ws + 917504;
  __bf16* ow1_bf  = ws + 1179648;

  auto cvt = [&](const float* s, __bf16* d, int n) {
    cvt_f32_bf16<<<(n + 255) / 256, 256, 0, stream>>>(s, d, n);
  };
  cvt(enc_w2,  w2_bf,   65536);
  cvt(enc_w3,  w3_bf,   65536);
  cvt(hinit_w, hi_bf,  131072);
  cvt(cinit_w, ci_bf,  131072);
  cvt(w_hh0,   whh0_bf, 262144);
  cvt(w_ih1,   wih1_bf, 262144);
  cvt(w_hh1,   whh1_bf, 262144);
  cvt(out_w1,  ow1_bf,  32768);

  (void)hipFuncSetAttribute((const void*)traj_main,
                            hipFuncAttributeMaxDynamicSharedMemorySize, LDS_BYTES);

  traj_main<<<8192 / ROWS, NTHREADS, LDS_BYTES, stream>>>(
      x, enc_w1, enc_b1, enc_b2, enc_b3, hinit_b, cinit_b,
      w_ih0, b_ih0, b_hh0, b_ih1, b_hh1, out_b1, out_w2, out_b2, start,
      w2_bf, w3_bf, hi_bf, ci_bf, whh0_bf, wih1_bf, whh1_bf, ow1_bf,
      (float*)d_out);
}
